// NONLocalBlock2D_489626272347
// MI455X (gfx1250) — matrix operand order repaired and checked by reading, not compiled, not run on hardware
//
#include <hip/hip_runtime.h>
#include <hip/hip_bf16.h>
#include <cstdint>

// NonLocal block, fused flash-attention style, bf16 WMMA on gfx1250 (wave32).
// Shapes: B=4, C=256, N=H*W=4096, IC=32.

typedef __attribute__((ext_vector_type(16))) __bf16 v16bf;
typedef __attribute__((ext_vector_type(8)))  __bf16 v8bf;
typedef __attribute__((ext_vector_type(8)))  float  v8f;

#define BATCH 4
#define CIN   256
#define NPIX  4096
#define ICH   32

__device__ __forceinline__ v8f wmma_bf16(v16bf a, v16bf b, v8f c) {
    // D = A(16x32 bf16) * B(32x16 bf16) + C(16x16 f32)
    return __builtin_amdgcn_wmma_f32_16x16x32_bf16(
        /*neg_a=*/false, a, /*neg_b=*/false, b,
        /*c_mod=*/(short)0, c, /*reuse_a=*/false, /*reuse_b=*/false);
}

// XOR-16 lane swap via DS_SWIZZLE group-of-32 mode (and=0x1f, or=0, xor=0x10):
// one DS op, no index VGPR / clamp arithmetic like __shfl_xor generates.
__device__ __forceinline__ float swz16(float x) {
    return __int_as_float(__builtin_amdgcn_ds_swizzle(__float_as_int(x), 0x401f));
}

// Assemble a 16x32 A-matrix operand from a 64B row: lane(m,h) takes row bytes
// [32h,32h+32)  (K order used here: VGPR v <-> K=16h+2v)
__device__ __forceinline__ v16bf a_from_row(const __bf16* row, int h) {
    const v8bf lo = *(const v8bf*)(row + 16 * h);
    const v8bf hi = *(const v8bf*)(row + 16 * h + 8);
    v16bf a;
#pragma unroll
    for (int i = 0; i < 8; ++i) { a[i] = lo[i]; a[8 + i] = hi[i]; }
    return a;
}

// ---------------------------------------------------------------------------
// Kernel 1: channel projections th = theta_w @ x, gx = g_w @ x (per pixel).
//   th  stored row-major bf16 [B][N][IC]
//   gxT stored transposed bf16 [B][IC][N]
// ---------------------------------------------------------------------------
__global__ __launch_bounds__(256)
void nlb_proj(const float* __restrict__ x, const float* __restrict__ g_w,
              const float* __restrict__ theta_w,
              __bf16* __restrict__ th, __bf16* __restrict__ gxT) {
    __shared__ float wt[CIN * ICH];  // theta_w transposed: [c][i]
    __shared__ float wg[CIN * ICH];  // g_w transposed:     [c][i]
    for (int t = threadIdx.x; t < CIN * ICH; t += 256) {
        const int i = t >> 8;
        const int c = t & 255;
        wt[c * ICH + i] = theta_w[t];
        wg[c * ICH + i] = g_w[t];
    }
    __syncthreads();

    const int b = blockIdx.x >> 4;
    const int n = ((blockIdx.x & 15) << 8) + threadIdx.x;
    const float* xp = x + (size_t)b * CIN * NPIX + n;

    float at[ICH], ag[ICH];
#pragma unroll
    for (int i = 0; i < ICH; ++i) { at[i] = 0.f; ag[i] = 0.f; }

    for (int c = 0; c < CIN; ++c) {
        const float xv = xp[(size_t)c * NPIX];           // coalesced across lanes
        const float4* wtc = (const float4*)(wt + c * ICH);
        const float4* wgc = (const float4*)(wg + c * ICH);
#pragma unroll
        for (int q = 0; q < 8; ++q) {
            const float4 a = wtc[q];
            const float4 g = wgc[q];
            at[4*q+0] = fmaf(a.x, xv, at[4*q+0]);
            at[4*q+1] = fmaf(a.y, xv, at[4*q+1]);
            at[4*q+2] = fmaf(a.z, xv, at[4*q+2]);
            at[4*q+3] = fmaf(a.w, xv, at[4*q+3]);
            ag[4*q+0] = fmaf(g.x, xv, ag[4*q+0]);
            ag[4*q+1] = fmaf(g.y, xv, ag[4*q+1]);
            ag[4*q+2] = fmaf(g.z, xv, ag[4*q+2]);
            ag[4*q+3] = fmaf(g.w, xv, ag[4*q+3]);
        }
    }

    const size_t nb = (size_t)b * NPIX + n;
    v16bf t0, t1;
#pragma unroll
    for (int i = 0; i < 16; ++i) { t0[i] = (__bf16)at[i]; t1[i] = (__bf16)at[16 + i]; }
    *(v16bf*)(th + nb * ICH)      = t0;
    *(v16bf*)(th + nb * ICH + 16) = t1;

    __bf16* gp = gxT + (size_t)b * ICH * NPIX + n;
#pragma unroll
    for (int i = 0; i < ICH; ++i) gp[(size_t)i * NPIX] = (__bf16)ag[i];
}

// ---------------------------------------------------------------------------
// Kernel 2: flash attention, transposed formulation. 8 waves/block, one
// 16-query tile per wave. Per 32-key chunk (double-buffered in LDS via
// GLOBAL_LOAD_ASYNC_TO_LDS_B128 / ASYNCcnt, issued by wave 0):
//   S^T = K * Q^T   : 2 WMMA (keys on M axis, queries on lanes)
//   y^T += G^T * P^T: 2 WMMA
// Keys-on-rows means softmax max/sum are lane-local over 16 registers plus a
// single ds_swizzle(xor16); the P^T relayout is an 8-value partner exchange —
// no LDS round trip, no butterfly chains.
// ---------------------------------------------------------------------------
__global__ __launch_bounds__(256)
void nlb_attn(const __bf16* __restrict__ th, const __bf16* __restrict__ gxT,
              __bf16* __restrict__ y) {
    __shared__ __align__(64) __bf16 kbuf[2 * 32 * ICH];   // [buf][key][ic]
    __shared__ __align__(64) __bf16 vbuf[2 * ICH * 32];   // [buf][ic][key]

    const int tid  = threadIdx.x;
    const int w    = tid >> 5;
    const int lane = tid & 31;
    const int h    = lane >> 4;
    const int l16  = lane & 15;
    const int b    = blockIdx.x >> 5;                      // 32 blocks per batch
    const int qbase = ((((blockIdx.x & 31) << 3) + w) << 4);

    const __bf16* thb = th  + (size_t)b * NPIX * ICH;
    const __bf16* gb  = gxT + (size_t)b * ICH * NPIX;

    const unsigned kbase = (unsigned)(uintptr_t)kbuf;
    const unsigned vbase = (unsigned)(uintptr_t)vbuf;

    // Async-copy lane addressing (wave 0 only). K chunk: 2KB contiguous;
    // V chunk: 32 rows x 64B, global row stride 8192B; per-inst addresses
    // precomputed so inst offset t*512 steps global and LDS consistently.
    const unsigned kva   = lane * 16u;
    const unsigned kdst0 = kbase + kva;
    const unsigned vdst0 = vbase + (lane >> 2) * 64u + (lane & 3) * 16u;
    unsigned va_t[4];
#pragma unroll
    for (int t = 0; t < 4; ++t)
        va_t[t] = (unsigned)((t * 8 + (lane >> 2)) * (NPIX * 2) + (lane & 3) * 16 - t * 512);

    // Prologue: stage chunk 0 into buffer 0
    if (tid < 32) {
        const unsigned long long ksrc = (unsigned long long)(uintptr_t)thb;
        const unsigned long long vsrc = (unsigned long long)(uintptr_t)gb;
        asm volatile(
            "global_load_async_to_lds_b128 %0, %1, %2 offset:0\n\t"
            "global_load_async_to_lds_b128 %0, %1, %2 offset:512\n\t"
            "global_load_async_to_lds_b128 %0, %1, %2 offset:1024\n\t"
            "global_load_async_to_lds_b128 %0, %1, %2 offset:1536"
            :: "v"(kdst0), "v"(kva), "s"(ksrc) : "memory");
        asm volatile(
            "global_load_async_to_lds_b128 %0, %1, %5 offset:0\n\t"
            "global_load_async_to_lds_b128 %0, %2, %5 offset:512\n\t"
            "global_load_async_to_lds_b128 %0, %3, %5 offset:1024\n\t"
            "global_load_async_to_lds_b128 %0, %4, %5 offset:1536"
            :: "v"(vdst0), "v"(va_t[0]), "v"(va_t[1]), "v"(va_t[2]), "v"(va_t[3]),
               "s"(vsrc) : "memory");
        asm volatile("s_wait_asynccnt 0x0" ::: "memory");
    }
    __syncthreads();

    // Q^T as loop-invariant B operand: lane(n=query l16,h): ic [16h,16h+16)
    const v16bf qb = *(const v16bf*)(thb + (size_t)(qbase + l16) * ICH + 16 * h);

    float m = -3.0e38f, l = 0.f;
    v8f o0 = {}, o1 = {};   // y^T tiles: ic rows [0,16) and [16,32), query cols

    for (int jj = 0; jj < NPIX; jj += 32) {
        const int buf = (jj >> 5) & 1;

        // Wave 0: kick off async staging of the next chunk into the other buffer
        if (tid < 32 && (jj + 32) < NPIX) {
            const int nb = buf ^ 1;
            const unsigned kd = kdst0 + nb * 2048u;
            const unsigned vd = vdst0 + nb * 2048u;
            const unsigned long long ksrc =
                (unsigned long long)(uintptr_t)thb + (unsigned long long)(jj + 32) * 64ull;
            const unsigned long long vsrc =
                (unsigned long long)(uintptr_t)gb + (unsigned long long)(jj + 32) * 2ull;
            asm volatile(
                "global_load_async_to_lds_b128 %0, %1, %2 offset:0\n\t"
                "global_load_async_to_lds_b128 %0, %1, %2 offset:512\n\t"
                "global_load_async_to_lds_b128 %0, %1, %2 offset:1024\n\t"
                "global_load_async_to_lds_b128 %0, %1, %2 offset:1536"
                :: "v"(kd), "v"(kva), "s"(ksrc) : "memory");
            asm volatile(
                "global_load_async_to_lds_b128 %0, %1, %5 offset:0\n\t"
                "global_load_async_to_lds_b128 %0, %2, %5 offset:512\n\t"
                "global_load_async_to_lds_b128 %0, %3, %5 offset:1024\n\t"
                "global_load_async_to_lds_b128 %0, %4, %5 offset:1536"
                :: "v"(vd), "v"(va_t[0]), "v"(va_t[1]), "v"(va_t[2]), "v"(va_t[3]),
                   "s"(vsrc) : "memory");
        }

        const __bf16* kb = kbuf + buf * 1024;
        const __bf16* vb = vbuf + buf * 1024;

        // S^T = K * Q^T : A tiles = key rows (A-layout) from LDS K buffer
        const v16bf ka0 = a_from_row(kb + l16 * ICH, h);          // keys jj+0..15
        const v16bf ka1 = a_from_row(kb + (16 + l16) * ICH, h);   // keys jj+16..31
        const v8f z = {};
        v8f s0 = wmma_bf16(ka0, qb, z);   // lane holds keys {jj+v+8h}, query l16
        v8f s1 = wmma_bf16(ka1, qb, z);   // lane holds keys {jj+16+v+8h}

        // Lane-local softmax over this lane's 16 scores + one partner combine
        float mx = s0[0];
#pragma unroll
        for (int v = 1; v < 8; ++v) mx = fmaxf(mx, s0[v]);
#pragma unroll
        for (int v = 0; v < 8; ++v) mx = fmaxf(mx, s1[v]);
        mx = fmaxf(mx, swz16(mx));
        const float mn = fmaxf(m, mx);
        const float sc = __expf(m - mn);
        m = mn;
#pragma unroll
        for (int v = 0; v < 8; ++v) { s0[v] = __expf(s0[v] - mn); s1[v] = __expf(s1[v] - mn); }
        float rs = 0.f;
#pragma unroll
        for (int v = 0; v < 8; ++v) rs += s0[v] + s1[v];
        rs += swz16(rs);
        l = l * sc + rs;
#pragma unroll
        for (int v = 0; v < 8; ++v) { o0[v] *= sc; o1[v] *= sc; }

        // Assemble P^T as B'-layout bf16: exchange 8 values with xor-16 partner.
        // h=0 lane sends its s1 rows (receives partner s0 = chunk-keys 8..15);
        // h=1 lane sends its s0 rows (receives partner s1 = chunk-keys 16..23).
        float rcv[8];
#pragma unroll
        for (int v = 0; v < 8; ++v) rcv[v] = swz16(h ? s0[v] : s1[v]);
        v16bf pab;
#pragma unroll
        for (int v = 0; v < 4; ++v) {       // K-pairs 16h+2v,+1 -> chunk-keys 16h+2v
            pab[2 * v]     = (__bf16)(h ? rcv[2 * v]     : s0[2 * v]);
            pab[2 * v + 1] = (__bf16)(h ? rcv[2 * v + 1] : s0[2 * v + 1]);
        }
#pragma unroll
        for (int v = 4; v < 8; ++v) {       // chunk-keys 16h+8+2(v-4)
            const int u = 2 * (v - 4);
            pab[2 * v]     = (__bf16)(h ? s1[u]     : rcv[u]);
            pab[2 * v + 1] = (__bf16)(h ? s1[u + 1] : rcv[u + 1]);
        }

        // y^T += G^T * P^T : A' tiles = gxT ic rows (A-layout) from LDS V buffer
        const v16bf va0 = a_from_row(vb + l16 * 32, h);          // ic 0..15
        const v16bf va1 = a_from_row(vb + (16 + l16) * 32, h);   // ic 16..31
        o0 = wmma_bf16(va0, pab, o0);
        o1 = wmma_bf16(va1, pab, o1);

        // Ensure next chunk landed in LDS before anyone reads it next iteration
        if (tid < 32) asm volatile("s_wait_asynccnt 0x0" ::: "memory");
        __syncthreads();
    }

    // Normalize, store y[b][q][ic]: lane's o0 -> ic [8h,8h+8), o1 -> ic [16+8h,+8)
    const float inv = 1.f / l;
    __bf16* yrow = y + ((size_t)b * NPIX + qbase + l16) * ICH;
    v8bf w0, w1;
#pragma unroll
    for (int v = 0; v < 8; ++v) { w0[v] = (__bf16)(o0[v] * inv); w1[v] = (__bf16)(o1[v] * inv); }
    *(v8bf*)(yrow + 8 * h)      = w0;
    *(v8bf*)(yrow + 16 + 8 * h) = w1;
}

// ---------------------------------------------------------------------------
// Kernel 3: out = W_w @ y^T + x.  One wave per (b, c-tile, 128-pixel group); 8 n-tiles each.
//   K = IC = 32 -> one WMMA per 16x16 output tile; residual in f32.
// ---------------------------------------------------------------------------
__global__ __launch_bounds__(32)
void nlb_out(const float* __restrict__ x, const float* __restrict__ W_w,
             const __bf16* __restrict__ y, float* __restrict__ out) {
    const int lane = threadIdx.x;
    const int h    = lane >> 4;
    const int l16  = lane & 15;
    int id = blockIdx.x;
    const int ng = id & 31; id >>= 5;
    const int ct = id & 15;
    const int b  = id >> 4;
    const int cbase = ct << 4;

    v16bf wa;
    {
        const float* wrow = W_w + (size_t)(cbase + l16) * ICH;
#pragma unroll
        for (int v = 0; v < 8; ++v) {
            const int k0 = 16 * h + 2 * v;
            wa[2 * v]     = (__bf16)wrow[k0];
            wa[2 * v + 1] = (__bf16)wrow[k0 + 1];
        }
    }

    const __bf16* yb = y + (size_t)b * NPIX * ICH;
    const float*  xb = x   + ((size_t)(b * CIN + cbase)) * NPIX;
    float*        ob = out + ((size_t)(b * CIN + cbase)) * NPIX;

#pragma unroll
    for (int t = 0; t < 8; ++t) {
        const int nbase = ng * 128 + t * 16;
        const v16bf by = *(const v16bf*)(yb + (size_t)(nbase + l16) * ICH + 16 * h);
        const v8f z = {};
        const v8f d = wmma_bf16(wa, by, z);
#pragma unroll
        for (int v = 0; v < 8; ++v) {
            const size_t off = (size_t)(v + 8 * h) * NPIX + nbase + l16;
            ob[off] = d[v] + xb[off];
        }
    }
}

// ---------------------------------------------------------------------------
extern "C" void kernel_launch(void* const* d_in, const int* in_sizes, int n_in,
                              void* d_out, int out_size, void* d_ws, size_t ws_size,
                              hipStream_t stream) {
    (void)in_sizes; (void)n_in; (void)out_size; (void)ws_size;
    const float* x       = (const float*)d_in[0];
    const float* g_w     = (const float*)d_in[1];
    const float* theta_w = (const float*)d_in[2];
    const float* W_w     = (const float*)d_in[3];
    float* out = (float*)d_out;

    // Workspace: th (1MB) | gxT (1MB) | y (1MB), all bf16
    const size_t seg = (size_t)BATCH * NPIX * ICH * sizeof(__bf16);
    __bf16* th  = (__bf16*)d_ws;
    __bf16* gxT = (__bf16*)((char*)d_ws + seg);
    __bf16* yb  = (__bf16*)((char*)d_ws + 2 * seg);

    nlb_proj<<<BATCH * (NPIX / 256), 256, 0, stream>>>(x, g_w, theta_w, th, gxT);
    nlb_attn<<<BATCH * 32, 256, 0, stream>>>(th, gxT, yb);
    nlb_out<<<BATCH * 16 * 32, 32, 0, stream>>>(x, W_w, yb, out);
}
